// TopKGate_532575945257
// MI455X (gfx1250) — matrix operand order repaired and checked by reading, not compiled, not run on hardware
//
#include <hip/hip_runtime.h>
#include <hip/hip_bf16.h>
#include <math.h>

#if defined(__HIP_DEVICE_COMPILE__) && !__has_builtin(__builtin_amdgcn_wmma_f32_16x16x32_bf16)
#error "gfx1250 bf16 WMMA builtin not available on this toolchain"
#endif

typedef __bf16 v4bf  __attribute__((ext_vector_type(4)));
typedef __bf16 v8bf  __attribute__((ext_vector_type(8)));
typedef __bf16 v16bf __attribute__((ext_vector_type(16)));
typedef float  v8f   __attribute__((ext_vector_type(8)));

#define S_TOK    8192
#define DDIM     4096
#define NEXP     64
#define CAP      128
#define BLK_ROWS 128
#define KT       32                   // one 16x16x32 WMMA K-step per chunk
#define NCHUNK   (DDIM / KT)          // 128
#define LDK      40                   // padded row stride in bf16 elements
#define GSTR     65                   // gates row stride (f32, odd -> conflict-free)
#define NBLK     (S_TOK / BLK_ROWS)   // 64
#define SEC      ((long long)S_TOK * NEXP * CAP)  // 67108864

// LDS pool byte offsets (bf16 hi/lo planes, double buffered)
#define AHI_OFF(b) ((b) * 20480)            // 128*LDK bf16 = 10240 B each
#define ALO_OFF(b) (10240 + (b) * 20480)
#define BHI_OFF(b) (40960 + (b) * 10240)    // 64*LDK bf16 = 5120 B each
#define BLO_OFF(b) (46080 + (b) * 10240)
#define POOL_BYTES 61504                    // >= 61440 stage, >= 33536 epilogue alias

__device__ __forceinline__ v16bf bf16cat(v8bf a, v8bf b) {
  return __builtin_shufflevector(a, b, 0,1,2,3,4,5,6,7,8,9,10,11,12,13,14,15);
}

// ---------------------------------------------------------------------------
// Kernel 1: fused logits GEMM (bf16x3 split-precision WMMA) + softmax/argmax
// ---------------------------------------------------------------------------
__global__ __launch_bounds__(256) void moe_gate_gemm(
    const float* __restrict__ x, const float* __restrict__ wg,
    int* __restrict__ idx_ws, float* __restrict__ gate_ws,
    float* __restrict__ gsum_ws, int* __restrict__ cnt_ws)
{
  __shared__ __align__(16) unsigned char poolraw[POOL_BYTES];

  const int tid  = threadIdx.x;
  const int lane = tid & 31;
  const int wv   = tid >> 5;        // wave 0..7 -> rows wv*16..wv*16+15
  const int l15  = lane & 15;
  const int half = lane >> 4;       // A: K {0-7,16-23} vs {8-15,24-31}; C: M 0-7 vs 8-15
  const int rowBase = blockIdx.x * BLK_ROWS;

  v8f acc0 = {}, acc1 = {}, acc2 = {}, acc3 = {};

  // stage one KT-chunk: global f32 -> split (hi,lo) bf16 planes in LDS
  auto stage = [&](int c, int buf) {
    const int kb = c * KT;
    __bf16* ah = (__bf16*)(poolraw + AHI_OFF(buf));
    __bf16* al = (__bf16*)(poolraw + ALO_OFF(buf));
    __bf16* bh = (__bf16*)(poolraw + BHI_OFF(buf));
    __bf16* bl = (__bf16*)(poolraw + BLO_OFF(buf));
#pragma unroll
    for (int p = 0; p < 4; ++p) {           // A: 128 rows x 32 k = 1024 float4
      int li = p * 256 + tid;
      int row = li >> 3, c4 = li & 7;
      const float4 v = *(const float4*)(x + (size_t)(rowBase + row) * DDIM + kb + c4 * 4);
      v4bf h, l;
      h[0]=(__bf16)v.x; l[0]=(__bf16)(v.x-(float)h[0]);
      h[1]=(__bf16)v.y; l[1]=(__bf16)(v.y-(float)h[1]);
      h[2]=(__bf16)v.z; l[2]=(__bf16)(v.z-(float)h[2]);
      h[3]=(__bf16)v.w; l[3]=(__bf16)(v.w-(float)h[3]);
      *(v4bf*)&ah[row * LDK + c4 * 4] = h;
      *(v4bf*)&al[row * LDK + c4 * 4] = l;
    }
#pragma unroll
    for (int p = 0; p < 2; ++p) {           // B: 64 rows x 32 k = 512 float4
      int li = p * 256 + tid;
      int row = li >> 3, c4 = li & 7;
      const float4 v = *(const float4*)(wg + (size_t)row * DDIM + kb + c4 * 4);
      v4bf h, l;
      h[0]=(__bf16)v.x; l[0]=(__bf16)(v.x-(float)h[0]);
      h[1]=(__bf16)v.y; l[1]=(__bf16)(v.y-(float)h[1]);
      h[2]=(__bf16)v.z; l[2]=(__bf16)(v.z-(float)h[2]);
      h[3]=(__bf16)v.w; l[3]=(__bf16)(v.w-(float)h[3]);
      *(v4bf*)&bh[row * LDK + c4 * 4] = h;
      *(v4bf*)&bl[row * LDK + c4 * 4] = l;
    }
  };

  stage(0, 0);
  __syncthreads();

  for (int c = 0; c < NCHUNK; ++c) {
    if (c + 1 < NCHUNK) stage(c + 1, (c + 1) & 1);
    if (c + 2 < NCHUNK)
      __builtin_prefetch(x + (size_t)(rowBase + (tid >> 1)) * DDIM + (size_t)(c + 2) * KT, 0, 0);

    const int buf = c & 1;
    const __bf16* Ah = (const __bf16*)(poolraw + AHI_OFF(buf));
    const __bf16* Al = (const __bf16*)(poolraw + ALO_OFF(buf));
    const __bf16* Bh = (const __bf16*)(poolraw + BHI_OFF(buf));
    const __bf16* Bl = (const __bf16*)(poolraw + BLO_OFF(buf));

    // A operand (16-bit A 16x32 layout): lane half 0 -> K 0-7 & 16-23,
    // lane half 1 -> K 8-15 & 24-31; two 8-elem contiguous runs each.
    const int arow = wv * 16 + l15;
    v8bf ah0 = *(const v8bf*)(Ah + arow * LDK + 8 * half);
    v8bf ah1 = *(const v8bf*)(Ah + arow * LDK + 16 + 8 * half);
    v8bf al0 = *(const v8bf*)(Al + arow * LDK + 8 * half);
    v8bf al1 = *(const v8bf*)(Al + arow * LDK + 16 + 8 * half);
    v16bf a_hi = bf16cat(ah0, ah1);
    v16bf a_lo = bf16cat(al0, al1);

#pragma unroll
    for (int t = 0; t < 4; ++t) {
      // B operand (32x16): lane half 0 -> K 0-7 & 16-23, half 1 -> K 8-15 & 24-31
      const int brow = t * 16 + l15;
      v16bf b_hi = bf16cat(*(const v8bf*)(Bh + brow * LDK + 8 * half), *(const v8bf*)(Bh + brow * LDK + 16 + 8 * half));
      v16bf b_lo = bf16cat(*(const v8bf*)(Bl + brow * LDK + 8 * half), *(const v8bf*)(Bl + brow * LDK + 16 + 8 * half));
      v8f* acc = (t == 0) ? &acc0 : (t == 1) ? &acc1 : (t == 2) ? &acc2 : &acc3;
      *acc = __builtin_amdgcn_wmma_f32_16x16x32_bf16(false, a_hi, false, b_hi, (short)0, *acc, false, false);
      *acc = __builtin_amdgcn_wmma_f32_16x16x32_bf16(false, a_hi, false, b_lo, (short)0, *acc, false, false);
      *acc = __builtin_amdgcn_wmma_f32_16x16x32_bf16(false, a_lo, false, b_hi, (short)0, *acc, false, false);
    }
    __syncthreads();
  }

  // ---- epilogue: spill logits, softmax/argmax per token, block stats ----
  float* lG   = (float*)poolraw;            // [128][65] f32, aliases stage bufs
  int*   lCnt = (int*)(poolraw + 33280);    // [64]

#pragma unroll
  for (int j = 0; j < 8; ++j) {      // C layout: lanes0-15 M=j, lanes16-31 M=8+j
    int m = wv * 16 + half * 8 + j;
    lG[m * GSTR +  0 + l15] = acc0[j];
    lG[m * GSTR + 16 + l15] = acc1[j];
    lG[m * GSTR + 32 + l15] = acc2[j];
    lG[m * GSTR + 48 + l15] = acc3[j];
  }
  if (tid < NEXP) lCnt[tid] = 0;
  __syncthreads();

  if (tid < BLK_ROWS) {
    float* row = &lG[tid * GSTR];
    float mx = row[0]; int am = 0;
    for (int e = 1; e < NEXP; ++e) { float v = row[e]; if (v > mx) { mx = v; am = e; } }
    float sum = 0.f;
    for (int e = 0; e < NEXP; ++e) { float v = __expf(row[e] - mx); row[e] = v; sum += v; }
    float inv = 1.0f / sum;
    for (int e = 0; e < NEXP; ++e) row[e] *= inv;
    int s = rowBase + tid;
    idx_ws[s]  = am;
    gate_ws[s] = inv;                // top-1 gate prob = exp(0)/sum
    atomicAdd(&lCnt[am], 1);         // integer -> deterministic
  }
  __syncthreads();

  if (tid < NEXP) {                  // fixed-order column reduction (deterministic)
    float g = 0.f;
    for (int r = 0; r < BLK_ROWS; ++r) g += lG[r * GSTR + tid];
    gsum_ws[blockIdx.x * NEXP + tid] = g;
    cnt_ws[blockIdx.x * NEXP + tid]  = lCnt[tid];
  }
}

// ---------------------------------------------------------------------------
// Kernel 2: per-expert prefix over block histograms, exp_counts, l_aux
// ---------------------------------------------------------------------------
__global__ __launch_bounds__(64) void moe_gate_scan(
    const int* __restrict__ cnt_ws, const float* __restrict__ gsum_ws,
    int* __restrict__ boff_ws, float* __restrict__ out)
{
  __shared__ float prod[NEXP];
  const int e = threadIdx.x;
  int tot = 0;
  for (int b = 0; b < NBLK; ++b) {            // serial, deterministic
    boff_ws[b * NEXP + e] = tot;
    tot += cnt_ws[b * NEXP + e];
  }
  float gs = 0.f;
  for (int b = 0; b < NBLK; ++b) gs += gsum_ws[b * NEXP + e];

  const long long countsOff = 1 + 2 * SEC;
  out[countsOff + e] = (float)tot;            // exp_counts (pre-capacity)

  float me = gs * (1.0f / (float)S_TOK);
  float ce = (float)tot * (1.0f / (float)S_TOK);
  prod[e] = me * ce;
  __syncthreads();
  if (e == 0) {
    float l = 0.f;
    for (int i = 0; i < NEXP; ++i) l += prod[i];
    out[0] = l * (float)NEXP;                 // l_aux
  }
}

// ---------------------------------------------------------------------------
// Kernel 3: capacity-limited slot assignment + sparse scatter
// ---------------------------------------------------------------------------
__global__ __launch_bounds__(128) void moe_gate_scatter(
    const int* __restrict__ idx_ws, const float* __restrict__ gate_ws,
    const int* __restrict__ boff_ws, float* __restrict__ out)
{
  __shared__ int   sIdx[BLK_ROWS];
  __shared__ float sGate[BLK_ROWS];
  const int t = threadIdx.x;
  const int s = blockIdx.x * BLK_ROWS + t;
  sIdx[t]  = idx_ws[s];
  sGate[t] = gate_ws[s];
  __syncthreads();

  const int my = sIdx[t];
  int rank = 0;
  for (int j = 0; j < t; ++j) rank += (sIdx[j] == my);   // in-order rank
  const int slot = boff_ws[blockIdx.x * NEXP + my] + rank;
  if (slot < CAP) {
    long long pos = (long long)s * (NEXP * CAP) + (long long)my * CAP + slot;
    out[1 + pos]       = sGate[t];   // combine_weights[s, e, c]
    out[1 + SEC + pos] = 1.0f;       // dispatch_mask[s, e, c]
  }
  // dropped tokens (slot >= CAP) leave all-zero rows, matching the reference
}

// ---------------------------------------------------------------------------
extern "C" void kernel_launch(void* const* d_in, const int* in_sizes, int n_in,
                              void* d_out, int out_size, void* d_ws, size_t ws_size,
                              hipStream_t stream)
{
  const float* x  = (const float*)d_in[0];   // [S, D] f32
  const float* wg = (const float*)d_in[1];   // [E, D] f32
  float* out = (float*)d_out;                // [1 | S*E*C | S*E*C | E] f32

  char* w = (char*)d_ws;
  int*   idx_ws  = (int*)   (w);                       // S ints
  float* gate_ws = (float*) (w + 32768);               // S floats
  float* gsum_ws = (float*) (w + 65536);               // NBLK*NEXP floats
  int*   cnt_ws  = (int*)   (w + 81920);               // NBLK*NEXP ints
  int*   boff_ws = (int*)   (w + 98304);               // NBLK*NEXP ints

  // Output is ~512 MB of mostly zeros: bulk-zero, then sparse scatter.
  (void)hipMemsetAsync(d_out, 0, (size_t)out_size * sizeof(float), stream);

  moe_gate_gemm   <<<NBLK, 256, 0, stream>>>(x, wg, idx_ws, gate_ws, gsum_ws, cnt_ws);
  moe_gate_scan   <<<1,    64,  0, stream>>>(cnt_ws, gsum_ws, boff_ws, out);
  moe_gate_scatter<<<NBLK, 128, 0, stream>>>(idx_ws, gate_ws, boff_ws, out);
}
